// GRUActorCritic_45372034514971
// MI455X (gfx1250) — matrix operand order repaired and checked by reading, not compiled, not run on hardware
//
#include <hip/hip_runtime.h>
#include <hip/hip_bf16.h>

// ---------------------------------------------------------------------------
// GRU actor-critic for MI455X (gfx1250, wave32, WMMA bf16 16x16x32).
// T=512, B=64, I=256, H=1024, O=64.
//
// Persistent-kernel design:
//   - 64 co-resident blocks (one per 16-wide h-column tile), 8 waves each
//     (4 batch M-tiles x 2-way K-split -> 2 waves/SIMD for WMMA overlap).
//   - Each block copies its private 120 KB packed weight slab (r/z/n gate
//     B-fragments, bf16) into LDS ONCE, then loops all 512 timesteps.
//   - Time loop is parity-unrolled so the h ping-pong pointers stay
//     kernarg-rooted -> A loads lower to global_load_b128 (not flat_load,
//     which would consume DScnt and contend with the LDS weight stream).
//   - Per step: K=1280 GEMM folded from [x_t | h_prev]; r/z gates use one
//     accumulator, n gate keeps x-part / h-part separate (for r * gh_n).
//     K-split halves the dependent WMMA chain; upper waves reduce partials
//     into lower waves via a 12 KB LDS buffer.
//   - One software grid barrier per step; release fence before signal,
//     agent-scope acquire fence (WGP$ invalidate) after.
//   - h kept in f32 (state) + bf16 (WMMA A fragments); feat=relu(h) stored
//     bf16 (non-temporal) for the WMMA head GEMM (w_pol & w_val = 80 cols).
// ---------------------------------------------------------------------------

#define T_STEPS 512
#define BATCH   64
#define IN_DIM  256
#define HID     1024
#define OUT_POL 64
#define NCHUNK  40            // (IN_DIM + HID) / 32
#define XCHUNK  8             // IN_DIM / 32
#define HCHUNK  32            // HID / 32
#define HSPLIT  12            // h-chunks in lower K-half (x:8 + h:12 = 20)
#define NBLOCKS (HID / 16)    // 64 persistent blocks
#define WSLAB   (3 * NCHUNK * 512)        // ushorts per block weight slab
#define RED_FLOATS (4 * 32 * 24)          // 4 waves x 32 lanes x 24 floats
#define LDS_BYTES  (WSLAB * 2 + RED_FLOATS * 4)   // 122880 + 12288 = 135168

typedef __attribute__((ext_vector_type(16))) __bf16 v16bf;
typedef __attribute__((ext_vector_type(8)))  __bf16 v8bf;
typedef __attribute__((ext_vector_type(8)))  float  v8f;
typedef __attribute__((ext_vector_type(4)))  unsigned int u32x4;

__device__ __forceinline__ unsigned short f32_to_bf16_rne(float f) {
    unsigned int u = __float_as_uint(f);
    unsigned int r = u + 0x7FFFu + ((u >> 16) & 1u);
    return (unsigned short)(r >> 16);
}

__device__ __forceinline__ v8f wmma_bf16(v16bf a, v16bf b, v8f c) {
    return __builtin_amdgcn_wmma_f32_16x16x32_bf16(
        false, a, false, b, (short)0, c, false, false);
}

__device__ __forceinline__ float sigmoidf_(float x) {
    return 1.0f / (1.0f + __expf(-x));
}

// Build A fragment (16x32 bf16): lane<16 -> K {0..15}, lane>=16 ->
// K {16..31}, row = 16*mtile + (lane&15).
__device__ __forceinline__ v16bf load_a(const unsigned short* abase, int khalf) {
    v16bf a = *(const v16bf*)(abase + khalf * 16);
    return a;
}

// ---------------------------------------------------------------------------
// Prep kernels
// ---------------------------------------------------------------------------

// wpack[ct(64)][gate(3)][chunk(40)][lane(32)][16 bf16]
// lane<16: W[nrow, kbase+0..15]; lane>=16: W[nrow, kbase+16..31]
// nrow = gate*1024 + 16*ct + (lane&15), kbase = 32*chunk (k<256 -> w_ih).
__global__ void pack_w_kernel(const float* __restrict__ w_ih,
                              const float* __restrict__ w_hh,
                              unsigned short* __restrict__ wpack) {
    int idx = blockIdx.x * blockDim.x + threadIdx.x;
    const int total = 64 * 3 * NCHUNK * 32;
    if (idx >= total) return;
    int lane = idx & 31;
    int c    = (idx >> 5) % NCHUNK;
    int gate = ((idx >> 5) / NCHUNK) % 3;
    int ct   = idx / (32 * NCHUNK * 3);
    int nrow  = gate * HID + ct * 16 + (lane & 15);
    int kbase = 32 * c + ((lane >> 4) << 4);
    unsigned short* dst = wpack + (size_t)idx * 16;
#pragma unroll
    for (int j = 0; j < 16; ++j) {
        int k = kbase + j;
        float v = (k < IN_DIM) ? w_ih[(size_t)nrow * IN_DIM + k]
                               : w_hh[(size_t)nrow * HID + (k - IN_DIM)];
        dst[j] = f32_to_bf16_rne(v);
    }
}

// Head matrix [w_pol(64); w_val(1); zeros(15)] = 80x1024 ->
// whpack[nt(5)][chunk(32)][lane(32)][16 bf16]
__global__ void pack_whead_kernel(const float* __restrict__ w_pol,
                                  const float* __restrict__ w_val,
                                  unsigned short* __restrict__ whpack) {
    int idx = blockIdx.x * blockDim.x + threadIdx.x;
    const int total = 5 * HCHUNK * 32;
    if (idx >= total) return;
    int lane = idx & 31;
    int c    = (idx >> 5) % HCHUNK;
    int nt   = idx / (32 * HCHUNK);
    int nrow  = nt * 16 + (lane & 15);
    int kbase = 32 * c + ((lane >> 4) << 4);
    unsigned short* dst = whpack + (size_t)idx * 16;
#pragma unroll
    for (int j = 0; j < 16; ++j) {
        int k = kbase + j;
        float v = 0.0f;
        if (nrow < OUT_POL)       v = w_pol[(size_t)nrow * HID + k];
        else if (nrow == OUT_POL) v = w_val[k];
        dst[j] = f32_to_bf16_rne(v);
    }
}

// bias[0..1023]=b_ih_r+b_hh_r, [1024..2047]=b_ih_z+b_hh_z,
// [2048..3071]=b_ih_n, [3072..4095]=b_hh_n.  bhead[0..79]. Also zero barrier.
__global__ void pack_bias_kernel(const float* __restrict__ b_ih,
                                 const float* __restrict__ b_hh,
                                 const float* __restrict__ b_pol,
                                 const float* __restrict__ b_val,
                                 float* __restrict__ bias,
                                 float* __restrict__ bhead,
                                 unsigned int* __restrict__ barrier_mem) {
    int idx = blockIdx.x * blockDim.x + threadIdx.x;
    if (idx < HID) {
        bias[idx]           = b_ih[idx] + b_hh[idx];
        bias[HID + idx]     = b_ih[HID + idx] + b_hh[HID + idx];
        bias[2 * HID + idx] = b_ih[2 * HID + idx];
        bias[3 * HID + idx] = b_hh[2 * HID + idx];
    } else if (idx < HID + 80) {
        int j = idx - HID;
        float v = 0.0f;
        if (j < OUT_POL)       v = b_pol[j];
        else if (j == OUT_POL) v = b_val[0];
        bhead[j] = v;
    } else if (idx < HID + 80 + 16) {
        barrier_mem[idx - (HID + 80)] = 0u;   // arrive-count + generation
    }
}

__global__ void conv_x_kernel(const float* __restrict__ x,
                              unsigned short* __restrict__ xbf, int n) {
    int idx = blockIdx.x * blockDim.x + threadIdx.x;
    if (idx < n) xbf[idx] = f32_to_bf16_rne(x[idx]);
}

__global__ void init_h_kernel(const float* __restrict__ h0,
                              float* __restrict__ hf,
                              unsigned short* __restrict__ hbf, int n) {
    int idx = blockIdx.x * blockDim.x + threadIdx.x;
    if (idx < n) {
        float v = h0[idx];
        hf[idx] = v;
        hbf[idx] = f32_to_bf16_rne(v);
    }
}

// ---------------------------------------------------------------------------
// Software grid barrier (all NBLOCKS blocks co-resident).
// ---------------------------------------------------------------------------
__device__ __forceinline__ void grid_sync(unsigned int* bar, unsigned int* gen,
                                          unsigned int step) {
    __syncthreads();
    if (threadIdx.x == 0) {
        __threadfence();  // release h writes to device scope
        unsigned int old = __hip_atomic_fetch_add(bar, 1u, __ATOMIC_ACQ_REL,
                                                  __HIP_MEMORY_SCOPE_AGENT);
        if (old == (unsigned int)(NBLOCKS - 1)) {
            __hip_atomic_store(bar, 0u, __ATOMIC_RELAXED,
                               __HIP_MEMORY_SCOPE_AGENT);
            __hip_atomic_fetch_add(gen, 1u, __ATOMIC_RELEASE,
                                   __HIP_MEMORY_SCOPE_AGENT);
        } else {
            while (__hip_atomic_load(gen, __ATOMIC_ACQUIRE,
                                     __HIP_MEMORY_SCOPE_AGENT) <= step) {
                __builtin_amdgcn_s_sleep(1);
            }
        }
    }
    __syncthreads();
    __threadfence();  // acquire: invalidate WGP$ so h loads see remote writes
}

// ---------------------------------------------------------------------------
// One GRU timestep (inlined twice for ping-pong parity: all pointers stay
// kernarg-rooted so A loads lower to global_load, not flat_load).
// ---------------------------------------------------------------------------
__device__ __forceinline__ void gru_step_body(
    int t,
    const unsigned short* __restrict__ xbf,
    const unsigned short* __restrict__ hbf_in,
    const float*          __restrict__ hf_in,
    float*                __restrict__ hf_out,
    unsigned short*       __restrict__ hbf_out,
    unsigned short*       __restrict__ featbf,
    const unsigned short* slane, float* red, int ridx,
    int ksp, int mtile, int khalf, int arow, int col,
    float bR_, float bZ_, float bIN_, float bHN_)
{
    unsigned short* feat_out = featbf + (size_t)t * BATCH * HID;
    const unsigned short* xrow = xbf + ((size_t)t * BATCH + arow) * IN_DIM;
    const unsigned short* hrow = hbf_in + (size_t)arow * HID;

    v8f accR = {}, accZ = {}, accNX = {}, accNH = {};

    if (ksp == 0) {
        __builtin_prefetch(xrow + BATCH * IN_DIM, 0, 1);  // next step's x
        // ---- x-side chunks (K 0..255) ----
#pragma unroll
        for (int c = 0; c < XCHUNK; ++c) {
            v16bf a = load_a(xrow + 32 * c, khalf);
            v16bf bR = *(const v16bf*)(slane + (size_t)(0 * NCHUNK + c) * 512);
            v16bf bZ = *(const v16bf*)(slane + (size_t)(1 * NCHUNK + c) * 512);
            v16bf bN = *(const v16bf*)(slane + (size_t)(2 * NCHUNK + c) * 512);
            accR  = wmma_bf16(a, bR, accR);
            accZ  = wmma_bf16(a, bZ, accZ);
            accNX = wmma_bf16(a, bN, accNX);
        }
        // ---- h-side chunks 0..HSPLIT-1 ----
#pragma unroll 4
        for (int c = 0; c < HSPLIT; ++c) {
            v16bf a = load_a(hrow + 32 * c, khalf);
            v16bf bR = *(const v16bf*)(slane + (size_t)(0 * NCHUNK + XCHUNK + c) * 512);
            v16bf bZ = *(const v16bf*)(slane + (size_t)(1 * NCHUNK + XCHUNK + c) * 512);
            v16bf bN = *(const v16bf*)(slane + (size_t)(2 * NCHUNK + XCHUNK + c) * 512);
            accR  = wmma_bf16(a, bR, accR);
            accZ  = wmma_bf16(a, bZ, accZ);
            accNH = wmma_bf16(a, bN, accNH);
        }
    } else {
        // ---- h-side chunks HSPLIT..31 ----
#pragma unroll 4
        for (int c = HSPLIT; c < HCHUNK; ++c) {
            v16bf a = load_a(hrow + 32 * c, khalf);
            v16bf bR = *(const v16bf*)(slane + (size_t)(0 * NCHUNK + XCHUNK + c) * 512);
            v16bf bZ = *(const v16bf*)(slane + (size_t)(1 * NCHUNK + XCHUNK + c) * 512);
            v16bf bN = *(const v16bf*)(slane + (size_t)(2 * NCHUNK + XCHUNK + c) * 512);
            accR  = wmma_bf16(a, bR, accR);
            accZ  = wmma_bf16(a, bZ, accZ);
            accNH = wmma_bf16(a, bN, accNH);
        }
        // publish partials for the lower K-half waves
#pragma unroll
        for (int j = 0; j < 8; ++j) {
            red[ridx + j]      = accR[j];
            red[ridx + 8 + j]  = accZ[j];
            red[ridx + 16 + j] = accNH[j];
        }
    }
    __syncthreads();

    if (ksp == 0) {
#pragma unroll
        for (int j = 0; j < 8; ++j) {
            accR[j]  += red[ridx + j];
            accZ[j]  += red[ridx + 8 + j];
            accNH[j] += red[ridx + 16 + j];
        }
        // ---- gates + state update ----
        // D layout: lane<16 -> (M=r, N=lane); lane>=16 -> (M=r+8, N=lane-16)
#pragma unroll
        for (int r = 0; r < 8; ++r) {
            const int m = (mtile << 4) + (khalf << 3) + r;
            float rg = sigmoidf_(accR[r] + bR_);
            float zg = sigmoidf_(accZ[r] + bZ_);
            float ng = tanhf(accNX[r] + bIN_ + rg * (accNH[r] + bHN_));
            float hp = hf_in[(size_t)m * HID + col];
            float hn = (1.0f - zg) * ng + zg * hp;
            hf_out[(size_t)m * HID + col]  = hn;
            hbf_out[(size_t)m * HID + col] = f32_to_bf16_rne(hn);
            __builtin_nontemporal_store(f32_to_bf16_rne(fmaxf(hn, 0.0f)),
                                        &feat_out[(size_t)m * HID + col]);
        }
    }
}

// ---------------------------------------------------------------------------
// Persistent GRU scan: 64 blocks x 256 threads (8 waves), whole sequence.
// Waves 0-3: M-tiles 0-3, K-chunks 0..19 (x + first 12 h chunks).
// Waves 4-7: M-tiles 0-3, K-chunks 20..39 (last 20 h chunks) -> LDS reduce.
// Dynamic LDS = 120 KB weights + 12 KB reduction (CDNA5: 320 KB/WGP).
// ---------------------------------------------------------------------------
__global__ __launch_bounds__(256) void gru_persistent_kernel(
    const unsigned short* __restrict__ xbf,     // [T,B,I] bf16
    const unsigned short* __restrict__ wpack,   // packed B fragments
    const float*          __restrict__ bias,    // 4*H combined biases
    float*          __restrict__ hf0,  float*          __restrict__ hf1,
    unsigned short* __restrict__ hbf0, unsigned short* __restrict__ hbf1,
    unsigned short* __restrict__ featbf,        // [T,B,H] relu(h) bf16
    unsigned int*   __restrict__ bar,
    unsigned int*   __restrict__ gen)
{
    extern __shared__ __align__(32) unsigned short sw[];  // WSLAB + reduction

    const int ct    = blockIdx.x;
    const int tid   = threadIdx.x;
    const int wave  = tid >> 5;          // 0..7
    const int mtile = wave & 3;          // batch tile
    const int ksp   = wave >> 2;         // K-split half
    const int lane  = tid & 31;
    const int lrow  = lane & 15;
    const int khalf = lane >> 4;

    // ---- Stage this block's weight slab into LDS once (contiguous copy) ----
    {
        const u32x4* src = (const u32x4*)(wpack + (size_t)ct * WSLAB);
        u32x4* dst = (u32x4*)sw;
#pragma unroll 4
        for (int u = tid; u < WSLAB / 8; u += 256) dst[u] = src[u];
    }
    float* red = (float*)(sw + WSLAB);   // [4 waves][32 lanes][24 floats]
    const int ridx = ((mtile << 5) + lane) * 24;
    __syncthreads();

    const int arow = (mtile << 4) + lrow;     // batch row fed by this lane (A)
    const int col  = (ct << 4) + lrow;        // h column owned by this lane (D)
    const float bR_  = bias[col];
    const float bZ_  = bias[HID + col];
    const float bIN_ = bias[2 * HID + col];
    const float bHN_ = bias[3 * HID + col];

    const unsigned short* slane = sw + (size_t)lane * 16;  // per-lane frag base

    // Parity-unrolled time loop: pointers stay kernarg-rooted (global loads).
    for (int t = 0; t < T_STEPS; t += 2) {
        gru_step_body(t, xbf, hbf0, hf0, hf1, hbf1, featbf,
                      slane, red, ridx, ksp, mtile, khalf, arow, col,
                      bR_, bZ_, bIN_, bHN_);
        grid_sync(bar, gen, (unsigned int)t);

        gru_step_body(t + 1, xbf, hbf1, hf1, hf0, hbf0, featbf,
                      slane, red, ridx, ksp, mtile, khalf, arow, col,
                      bR_, bZ_, bIN_, bHN_);
        grid_sync(bar, gen, (unsigned int)(t + 1));
    }
}

// ---------------------------------------------------------------------------
// Heads: logits + value from feat (bf16), 80-col packed weight matrix.
// 2048 blocks (M-tiles over T*B rows), 5 waves (4 logits N-tiles + 1 value).
// ---------------------------------------------------------------------------
__global__ __launch_bounds__(160) void heads_kernel(
    const unsigned short* __restrict__ featbf,
    const unsigned short* __restrict__ whpack,
    const float*          __restrict__ bhead,
    float* __restrict__ out_logits,             // [T*B, 64]
    float* __restrict__ out_val)                // [T*B]
{
    const int mt    = blockIdx.x;
    const int nt    = threadIdx.x >> 5;
    const int lane  = threadIdx.x & 31;
    const int lrow  = lane & 15;
    const int khalf = lane >> 4;

    v8f acc = {};
    const int arow = (mt << 4) + lrow;
    const unsigned short* abase0 = featbf + (size_t)arow * HID;

#pragma unroll 4
    for (int c = 0; c < HCHUNK; ++c) {
        v16bf a = load_a(abase0 + 32 * c, khalf);
        const unsigned short* bp =
            whpack + (((size_t)nt * HCHUNK + c) * 32 + lane) * 16;
        v16bf b = *(const v16bf*)bp;
        acc = wmma_bf16(a, b, acc);
    }

    const int col = (nt << 4) + lrow;
    const float bb = bhead[col];
#pragma unroll
    for (int r = 0; r < 8; ++r) {
        const int m = (mt << 4) + (khalf << 3) + r;
        float v = acc[r] + bb;
        if (col < OUT_POL)       out_logits[(size_t)m * OUT_POL + col] = v;
        else if (col == OUT_POL) out_val[m] = v;
    }
}

__global__ void copy_hlast_kernel(const float* __restrict__ hf,
                                  float* __restrict__ out, int n) {
    int idx = blockIdx.x * blockDim.x + threadIdx.x;
    if (idx < n) out[idx] = hf[idx];
}

// ---------------------------------------------------------------------------
// Host launcher
// ---------------------------------------------------------------------------
extern "C" void kernel_launch(void* const* d_in, const int* in_sizes, int n_in,
                              void* d_out, int out_size, void* d_ws, size_t ws_size,
                              hipStream_t stream) {
    (void)in_sizes; (void)n_in; (void)out_size; (void)ws_size;

    const float* x     = (const float*)d_in[0];
    const float* h0    = (const float*)d_in[1];
    const float* w_ih  = (const float*)d_in[2];
    const float* w_hh  = (const float*)d_in[3];
    const float* b_ih  = (const float*)d_in[4];
    const float* b_hh  = (const float*)d_in[5];
    const float* w_pol = (const float*)d_in[6];
    const float* b_pol = (const float*)d_in[7];
    const float* w_val = (const float*)d_in[8];
    const float* b_val = (const float*)d_in[9];

    char* ws = (char*)d_ws;
    size_t off = 0;
    unsigned short* wpack  = (unsigned short*)(ws + off); off += (size_t)64 * WSLAB * 2;
    unsigned short* whpack = (unsigned short*)(ws + off); off += (size_t)5 * HCHUNK * 32 * 16 * 2;
    unsigned short* xbf    = (unsigned short*)(ws + off); off += (size_t)T_STEPS * BATCH * IN_DIM * 2;
    unsigned short* featbf = (unsigned short*)(ws + off); off += (size_t)T_STEPS * BATCH * HID * 2;
    float* hf0             = (float*)(ws + off);          off += (size_t)BATCH * HID * 4;
    float* hf1             = (float*)(ws + off);          off += (size_t)BATCH * HID * 4;
    unsigned short* hbf0   = (unsigned short*)(ws + off); off += (size_t)BATCH * HID * 2;
    unsigned short* hbf1   = (unsigned short*)(ws + off); off += (size_t)BATCH * HID * 2;
    float* bias            = (float*)(ws + off);          off += (size_t)4 * HID * 4;
    float* bhead           = (float*)(ws + off);          off += 128 * 4;
    unsigned int* barmem   = (unsigned int*)(ws + off);   off += 16 * 4;   // [0]=count, [1]=gen

    float* out_logits = (float*)d_out;
    float* out_val    = out_logits + (size_t)T_STEPS * BATCH * OUT_POL;
    float* out_hlast  = out_val + (size_t)T_STEPS * BATCH;

    // --- prep ---
    {
        int n = 64 * 3 * NCHUNK * 32;
        pack_w_kernel<<<(n + 255) / 256, 256, 0, stream>>>(w_ih, w_hh, wpack);
    }
    {
        int n = 5 * HCHUNK * 32;
        pack_whead_kernel<<<(n + 255) / 256, 256, 0, stream>>>(w_pol, w_val, whpack);
    }
    pack_bias_kernel<<<(HID + 96 + 255) / 256, 256, 0, stream>>>(
        b_ih, b_hh, b_pol, b_val, bias, bhead, barmem);
    {
        int n = T_STEPS * BATCH * IN_DIM;
        conv_x_kernel<<<(n + 255) / 256, 256, 0, stream>>>(x, xbf, n);
    }
    {
        int n = BATCH * HID;
        init_h_kernel<<<(n + 255) / 256, 256, 0, stream>>>(h0, hf0, hbf0, n);
    }

    // --- persistent GRU scan: one launch, 132 KB dynamic LDS per block ---
    hipFuncSetAttribute((const void*)gru_persistent_kernel,
                        hipFuncAttributeMaxDynamicSharedMemorySize, LDS_BYTES);
    gru_persistent_kernel<<<NBLOCKS, 256, LDS_BYTES, stream>>>(
        xbf, wpack, bias, hf0, hf1, hbf0, hbf1, featbf,
        &barmem[0], &barmem[1]);

    // --- heads ---
    heads_kernel<<<(T_STEPS * BATCH) / 16, 160, 0, stream>>>(
        featbf, whpack, bhead, out_logits, out_val);

    // --- h_last (after 512 steps, final state is in buffer 0) ---
    {
        int n = BATCH * HID;
        copy_hlast_kernel<<<(n + 255) / 256, 256, 0, stream>>>(hf0, out_hlast, n);
    }
}
